// deepBSDEplus_91164975824931
// MI455X (gfx1250) — matrix operand order repaired and checked by reading, not compiled, not run on hardware
//
#include <hip/hip_runtime.h>
#include <hip/hip_bf16.h>

typedef _Float16 v16h __attribute__((ext_vector_type(16)));
typedef _Float16 v8h  __attribute__((ext_vector_type(8)));
typedef float    v8f  __attribute__((ext_vector_type(8)));

#define HID   128
#define OUTD  100
#define NSTEP 64
#define NCOLS 65
#define BTOT  4096   // 2*BATCH samples

// Native CDNA5 V_TANH_F32 when available (gfx1250 TRANS op -> co-executes
// with WMMA); fallback keeps the code compiling on older toolchains.
__device__ inline float fast_tanh(float x) {
#if __has_builtin(__builtin_amdgcn_tanhf)
  return __builtin_amdgcn_tanhf(x);
#else
  return tanhf(x);
#endif
}

// ---------------------------------------------------------------------------
// Prep: transpose + convert weights to f16 in workspace.
// ws layout (in halves): Wt2[128][128] @ 0, Wt3[128][128] @ 16384,
//                        Wt4[100][128] @ 32768   (total 91,136 bytes)
// Wt*[n][k] = W*[k][n]  so a WMMA B-fragment row (fixed output neuron n,
// contiguous K) is a contiguous 256-byte run.
// ---------------------------------------------------------------------------
__global__ void prep_weights(const float* __restrict__ W2,
                             const float* __restrict__ W3,
                             const float* __restrict__ W4,
                             _Float16* __restrict__ ws) {
  int i = blockIdx.x * blockDim.x + threadIdx.x;
  int stride = gridDim.x * blockDim.x;
  for (int idx = i; idx < HID * HID; idx += stride) {
    int k = idx / HID, n = idx % HID;
    ws[n * HID + k]         = (_Float16)W2[idx];
    ws[16384 + n * HID + k] = (_Float16)W3[idx];
  }
  for (int idx = i; idx < HID * OUTD; idx += stride) {
    int k = idx / OUTD, n = idx % OUTD;
    ws[32768 + n * HID + k] = (_Float16)W4[idx];
  }
}

// A-fragment (16-bit A 16x32 layout): lane holds row M=lane&15; element
// e[0..7] = K [kA, kA+8), e[8..15] = K [kA+16, kA+24), kA = k0 + 8*(lane>>4).
__device__ inline v16h ldA(const _Float16* row, int kA) {
  v8h lo = *(const v8h*)(row + kA);
  v8h hi = *(const v8h*)(row + kA + 16);
  v16h r;
#pragma unroll
  for (int i = 0; i < 8; ++i) { r[i] = lo[i]; r[i + 8] = hi[i]; }
  return r;
}

// B-fragment (16-bit B 32x16 layout): lane holds column N=lane&15; e[0..7] =
// K [kB, kB+8), e[8..15] = K [kB+16, kB+24), kB = k0 + 8*(lane>>4).
__device__ inline v16h ldB(const _Float16* row, int kB) {
  v8h lo = *(const v8h*)(row + kB);
  v8h hi = *(const v8h*)(row + kB + 16);
  v16h r;
#pragma unroll
  for (int i = 0; i < 8; ++i) { r[i] = lo[i]; r[i + 8] = hi[i]; }
  return r;
}

// ---------------------------------------------------------------------------
// One wave32 per 16-sample tile. 256 blocks x 32 threads.
// ---------------------------------------------------------------------------
__global__ __launch_bounds__(32)
void bsde_kernel(const float* __restrict__ W1, const float* __restrict__ b1,
                 const float* __restrict__ b2, const float* __restrict__ b3,
                 const float* __restrict__ b4,
                 const int*   __restrict__ sn, const float* __restrict__ ss,
                 const float* __restrict__ sdB, const float* __restrict__ tptr,
                 const _Float16* __restrict__ ws, float* __restrict__ out) {
  // ping-pong activation buffers: [buf][stream(h,h',h'')][sample][k], f16
  __shared__ __align__(16) _Float16 act[2][3][16][HID];   // 24 KB

  const int lane = threadIdx.x;       // 0..31
  const int m    = lane & 15;         // sample within tile (both halves alias)
  const int half = lane >> 4;
  const int s0   = blockIdx.x * 16;   // first sample of this tile

  const float t  = tptr[0];
  const float dt = t / (float)NSTEP;

  const _Float16* Wt2 = ws;
  const _Float16* Wt3 = ws + 16384;
  const _Float16* Wt4 = ws + 32768;

  float u = 0.f, ps = 0.f, pss = 0.f;

  // i = -1 : initial eval at column N (tk = t), producing u, ps, pss.
  // i >= 0 : carry update with dB[:, N-1-i]; then (except last) re-eval
  //          ps,pss at column N-1-i with tk = t - dt*i  (== reference scan).
#pragma unroll 1
  for (int i = -1; i < NSTEP; ++i) {
    const int   col = (i < 0) ? NSTEP : (NSTEP - 1 - i);
    const float tk  = (i < 0) ? t : (t - dt * (float)i);

    if (i >= 0) {
      float dB = sdB[(s0 + m) * NSTEP + col];      // same value in both halves
      u = u - ps * dB - pss * dt * 0.5f;
      if (i == NSTEP - 1) break;                   // last eval is discarded
    }

    const float sv = ss[(s0 + m) * NCOLS + col];   // identical across halves
    const int   nv = sn[(s0 + m) * NCOLS + col];

    // ---- layer 1 (2 -> 128), analytic derivs: z1' = W1[0][j], z1'' = 0 ----
    {
      const int j0 = half * 64;
#pragma unroll 4
      for (int jj = 0; jj < 64; ++jj) {
        int   j  = j0 + jj;
        float w0 = W1[j];            // W1[0][j]
        float w1 = W1[HID + j];      // W1[1][j]
        float z  = sv * w0 + tk * w1 + b1[j];
        float h  = fast_tanh(z);
        float g  = 1.f - h * h;
        float dh = g * w0;
        float ddh = -2.f * h * dh * w0;
        act[0][0][m][j] = (_Float16)h;
        act[0][1][m][j] = (_Float16)dh;
        act[0][2][m][j] = (_Float16)ddh;
      }
    }

    // ---- layers 2 & 3 (128 -> 128) via WMMA, 3 streams share B ----
    int pp = 0;
#pragma unroll
    for (int l = 0; l < 2; ++l) {
      const _Float16* Wt   = (l == 0) ? Wt2 : Wt3;
      const float*    bias = (l == 0) ? b2  : b3;
      const _Float16 (*ain)[16][HID] = act[pp];
      _Float16 (*aout)[16][HID]      = act[pp ^ 1];

#pragma unroll 1
      for (int nch = 0; nch < 8; ++nch) {
        const int ncol = nch * 16 + m;             // this lane's output neuron
        v8f acc_h = {}, acc_d = {}, acc_dd = {};
#pragma unroll
        for (int ks = 0; ks < 4; ++ks) {
          const int k0 = ks * 32;
          const int kA = k0 + half * 8;
          v16h a_h  = ldA(&ain[0][m][0], kA);
          v16h a_d  = ldA(&ain[1][m][0], kA);
          v16h a_dd = ldA(&ain[2][m][0], kA);
          v16h bf   = ldB(Wt + ncol * HID, k0 + half * 8);
          acc_h  = __builtin_amdgcn_wmma_f32_16x16x32_f16(false, a_h,  false, bf,
                                                          (short)0, acc_h,  false, false);
          acc_d  = __builtin_amdgcn_wmma_f32_16x16x32_f16(false, a_d,  false, bf,
                                                          (short)0, acc_d,  false, false);
          acc_dd = __builtin_amdgcn_wmma_f32_16x16x32_f16(false, a_dd, false, bf,
                                                          (short)0, acc_dd, false, false);
        }
        // epilogue: C layout lane L, vgpr v -> (row = v + 8*half, col = ncol)
        const float bv = bias[ncol];
#pragma unroll
        for (int v = 0; v < 8; ++v) {
          const int mo = v + 8 * half;
          float h   = fast_tanh(acc_h[v] + bv);
          float g   = 1.f - h * h;
          float dz  = acc_d[v];
          float ddz = acc_dd[v];
          float dh  = g * dz;
          float ddh = g * ddz - 2.f * h * dh * dz;
          aout[0][mo][ncol] = (_Float16)h;
          aout[1][mo][ncol] = (_Float16)dh;
          aout[2][mo][ncol] = (_Float16)ddh;
        }
      }
      pp ^= 1;
    }

    // ---- layer 4: dot h3/h3'/h3'' with W4 column n (per-sample gather) ----
    // Vectorized: 8 halves (b128) per load, each half-wave sums 64 K values.
    {
      const _Float16 (*ain)[16][HID] = act[pp];    // pp == 0 after two layers
      const _Float16* wrow = Wt4 + nv * HID;
      float sh = 0.f, sd = 0.f, sdd = 0.f;
      const int kb = half * 64;
#pragma unroll
      for (int c = 0; c < 8; ++c) {
        v8h wv  = *(const v8h*)(wrow + kb + c * 8);
        v8h hv  = *(const v8h*)(&ain[0][m][kb + c * 8]);
        v8h dv  = *(const v8h*)(&ain[1][m][kb + c * 8]);
        v8h ddv = *(const v8h*)(&ain[2][m][kb + c * 8]);
#pragma unroll
        for (int e = 0; e < 8; ++e) {
          float w = (float)wv[e];
          sh  += (float)hv[e]  * w;
          sd  += (float)dv[e]  * w;
          sdd += (float)ddv[e] * w;
        }
      }
      sh  += __shfl_xor(sh,  16, 32);
      sd  += __shfl_xor(sd,  16, 32);
      sdd += __shfl_xor(sdd, 16, 32);
      ps  = sd;
      pss = sdd;
      if (i < 0) u = sh + b4[nv];
    }
  }

  // ---- outputs: u (4096) then u0 (4096) ----
  if (lane < 16) {
    out[s0 + m] = u;
    float sv0 = ss[(s0 + m) * NCOLS + 0];
    const float beta = 1000000.0f;
    float u0 = sqrtf(beta / (2.0f * 3.14159265358979323846f)) *
               __expf(-beta * sv0 * sv0 * 0.5f) * (1.0f / (float)OUTD);
    out[BTOT + s0 + m] = u0;
  }
}

extern "C" void kernel_launch(void* const* d_in, const int* in_sizes, int n_in,
                              void* d_out, int out_size, void* d_ws, size_t ws_size,
                              hipStream_t stream) {
  (void)in_sizes; (void)n_in; (void)out_size; (void)ws_size;
  const float* W1  = (const float*)d_in[0];
  const float* b1  = (const float*)d_in[1];
  const float* W2  = (const float*)d_in[2];
  const float* b2  = (const float*)d_in[3];
  const float* W3  = (const float*)d_in[4];
  const float* b3  = (const float*)d_in[5];
  const float* W4  = (const float*)d_in[6];
  const float* b4  = (const float*)d_in[7];
  const int*   sn  = (const int*)d_in[8];
  const float* ss  = (const float*)d_in[9];
  const float* sdB = (const float*)d_in[10];
  const float* t   = (const float*)d_in[11];
  _Float16* ws = (_Float16*)d_ws;
  float* out = (float*)d_out;

  prep_weights<<<64, 256, 0, stream>>>(W2, W3, W4, ws);
  bsde_kernel<<<BTOT / 16, 32, 0, stream>>>(W1, b1, b2, b3, b4,
                                            sn, ss, sdB, t, ws, out);
}
